// MANN_11364483465731
// MI455X (gfx1250) — matrix operand order repaired and checked by reading, not compiled, not run on hardware
//
#include <hip/hip_runtime.h>
#include <math.h>

typedef _Float16 v16h __attribute__((ext_vector_type(16)));
typedef _Float16 v8h  __attribute__((ext_vector_type(8)));
typedef float    v8f  __attribute__((ext_vector_type(8)));
typedef int      v4i  __attribute__((ext_vector_type(4)));

#define B_    1024
#define IN_   1024
#define OUT_  1024
#define H_    1024
#define GH_   512
#define E_    16
#define GI_   64
#define OUTW  (OUT_ + E_)   // 1040 output columns (m3 ++ ew)

// ---------------------------------------------------------------------------
// CDNA5 async copy global->LDS (ASYNCcnt path), with portable fallback.
// The builtin's params are address-space-qualified int4 pointers:
//   (global v4i* src, lds v4i* dst, imm offset, imm cpol)
// ---------------------------------------------------------------------------
#if __has_builtin(__builtin_amdgcn_global_load_async_to_lds_b128)
#define USE_ASYNC 1
#else
#define USE_ASYNC 0
#endif

__device__ __forceinline__ void cp_b128_to_lds(_Float16* lds_dst, const _Float16* g_src) {
#if USE_ASYNC
  __builtin_amdgcn_global_load_async_to_lds_b128(
      (__attribute__((address_space(1))) v4i*)g_src,
      (__attribute__((address_space(3))) v4i*)lds_dst,
      /*imm offset=*/0, /*cpol=*/0);
#else
  *(v8h*)lds_dst = *(const v8h*)g_src;
#endif
}

__device__ __forceinline__ void wait_async0() {
#if USE_ASYNC
#if __has_builtin(__builtin_amdgcn_s_wait_asynccnt)
  __builtin_amdgcn_s_wait_asynccnt(0);
#else
  asm volatile("s_wait_asynccnt 0" ::: "memory");
#endif
#endif
}

// ---------------------------------------------------------------------------
// fp32 -> f16 conversion (grid-stride)
// ---------------------------------------------------------------------------
__global__ void k_cvt_f16(const float* __restrict__ src, _Float16* __restrict__ dst,
                          size_t n) {
  size_t i = (size_t)blockIdx.x * blockDim.x + threadIdx.x;
  size_t stride = (size_t)gridDim.x * blockDim.x;
  for (; i < n; i += stride) dst[i] = (_Float16)src[i];
}

// ---------------------------------------------------------------------------
// Gating layer 1: h1[b,j] = elu( sum_k g1_w[j,k] * x[b, idx[k]] + g1_b[j] )
// ---------------------------------------------------------------------------
__global__ void k_gate1(const float* __restrict__ x, const int* __restrict__ idx,
                        const float* __restrict__ w, const float* __restrict__ b,
                        float* __restrict__ h1) {
  int t = blockIdx.x * blockDim.x + threadIdx.x;
  if (t >= B_ * GH_) return;
  int bi = t / GH_, jj = t - bi * GH_;
  const float* xr = x + (size_t)bi * IN_;
  const float* wr = w + (size_t)jj * GI_;
  float s = b[jj];
#pragma unroll 8
  for (int k = 0; k < GI_; ++k) s += wr[k] * xr[idx[k]];
  h1[t] = s > 0.f ? s : expm1f(s);
}

// ---------------------------------------------------------------------------
// Gating layer 2: h2 = elu(g2_w @ h1 + g2_b), (B,512) x (512,512)
// ---------------------------------------------------------------------------
__global__ void k_gate2(const float* __restrict__ h1, const float* __restrict__ w,
                        const float* __restrict__ b, float* __restrict__ h2) {
  int t = blockIdx.x * blockDim.x + threadIdx.x;
  if (t >= B_ * GH_) return;
  int bi = t / GH_, jj = t - bi * GH_;
  const float4* xr = (const float4*)(h1 + (size_t)bi * GH_);
  const float4* wr = (const float4*)(w + (size_t)jj * GH_);
  float s = b[jj];
#pragma unroll 4
  for (int k = 0; k < GH_ / 4; ++k) {
    float4 a = xr[k], c = wr[k];
    s += a.x * c.x + a.y * c.y + a.z * c.z + a.w * c.w;
  }
  h2[t] = s > 0.f ? s : expm1f(s);
}

// ---------------------------------------------------------------------------
// Gating layer 3 + softmax. One block (16 threads) per batch row.
// ---------------------------------------------------------------------------
__global__ void k_gate3_softmax(const float* __restrict__ h2, const float* __restrict__ w,
                                const float* __restrict__ b, float* __restrict__ ew,
                                float* __restrict__ out) {
  int bi = blockIdx.x;
  int e  = threadIdx.x;            // 0..15
  const float4* xr = (const float4*)(h2 + (size_t)bi * GH_);
  const float4* wr = (const float4*)(w + (size_t)e * GH_);
  float s = b[e];
#pragma unroll 4
  for (int k = 0; k < GH_ / 4; ++k) {
    float4 a = xr[k], c = wr[k];
    s += a.x * c.x + a.y * c.y + a.z * c.z + a.w * c.w;
  }
  float m = s;
  for (int off = 8; off >= 1; off >>= 1) m = fmaxf(m, __shfl_xor(m, off, 16));
  float p = __expf(s - m);
  float sum = p;
  for (int off = 8; off >= 1; off >>= 1) sum += __shfl_xor(sum, off, 16);
  float wgt = p / sum;
  ew[bi * E_ + e] = wgt;
  out[(size_t)bi * OUTW + OUT_ + e] = wgt;
}

// ---------------------------------------------------------------------------
// Expert-mixture layer (WMMA + async LDS staging):
//   out[b,n] = act( sum_e ew[b,e] * ( sum_k act[b,k]*W[e,n,k] + bias[e,n] ) )
//
// Block = 256 thr = 8 waves. Block tile: 64(batch) x 64(N). K chunked by 64.
// A fragments (2 x v16h) live in registers per K-chunk and are reused across
// all 16 experts (mixing is linear, so per-chunk partials can be ew-scaled).
// W tiles (64n x 64k f16 = 8KB) are async-staged to LDS, double buffered.
// ---------------------------------------------------------------------------
__global__ __launch_bounds__(256) void k_moe_layer(
    const _Float16* __restrict__ act,    // (B, K) f16
    const _Float16* __restrict__ wgt,    // (E, N, K) f16, row-major
    const float*    __restrict__ bias,   // (E, N)
    const float*    __restrict__ ew,     // (B, E)
    _Float16*       __restrict__ out_h,  // (B, N) f16 (may be null)
    float*          __restrict__ out_f,  // final fp32 out, row stride OUTW (may be null)
    int K, int N, int apply_elu) {
  __shared__ float    s_ew[64 * E_];            // 4 KB
  __shared__ _Float16 s_w[2][64 * 64];          // 2 x 8 KB, double buffered

  const int b_blk = blockIdx.x * 64;
  const int n_blk = blockIdx.y * 64;
  const int tid   = threadIdx.x;

  // Stage the 64x16 ew tile once.
  for (int i = tid; i < 64 * E_; i += 256)
    s_ew[i] = ew[(size_t)(b_blk + i / E_) * E_ + (i % E_)];

  const int wv       = tid >> 5;       // wave 0..7
  const int lane     = tid & 31;
  const int m_sub    = wv >> 1;        // 0..3
  const int n_sub    = wv & 1;         // 0..1 (covers 32 cols = two 16-tiles)
  const int row_half = lane >> 4;      // lane >= 16
  const int l16      = lane & 15;

  const int b0   = b_blk + m_sub * 16;
  const int n_g0 = n_blk + n_sub * 32 + l16;     // first 16-col tile
  const int n_g1 = n_g0 + 16;                    // second 16-col tile

  // A source: row b0+l16; per 32-wide k-step: 16 contiguous halves at +16*row_half.
  const _Float16* arow = act + (size_t)(b0 + l16) * K + row_half * 16;

  // W staging assignment: thread t copies 2x16B of row (t>>2), units (t&3)*2,+1.
  const int st_row = tid >> 2;               // 0..63 (n within tile)
  const int st_ku  = (tid & 3) * 2;          // unit of 8 halves: 0,2,4,6
  const size_t wstride_e = (size_t)N * K;
  const _Float16* wg_base = wgt + (size_t)(n_blk + st_row) * K + st_ku * 8;

  const int n_chunks = K / 64;               // 16
  const int S = n_chunks * E_;               // 256 flattened (kc,e) stages

  // Prologue: stage (kc=0, e=0) into buffer 0.
  {
    const _Float16* g = wg_base;             // e=0, kc=0
    _Float16* d = &s_w[0][st_row * 64 + st_ku * 8];
    cp_b128_to_lds(d, g);
    cp_b128_to_lds(d + 8, g + 8);
  }

  v8f accm0 = {0.f, 0.f, 0.f, 0.f, 0.f, 0.f, 0.f, 0.f};
  v8f accm1 = {0.f, 0.f, 0.f, 0.f, 0.f, 0.f, 0.f, 0.f};
  const int rb = m_sub * 16 + row_half * 8;  // base row in 64-row block tile

  for (int kc = 0; kc < n_chunks; ++kc) {
    // Load this chunk's A fragments into registers (reused by all experts).
    v16h a0, a1;
    {
      const _Float16* ap = arow + kc * 64;
      v8h lo0 = *(const v8h*)(ap);
      v8h hi0 = *(const v8h*)(ap + 8);
      v8h lo1 = *(const v8h*)(ap + 32);
      v8h hi1 = *(const v8h*)(ap + 40);
#pragma unroll
      for (int i = 0; i < 8; ++i) {
        a0[i] = lo0[i]; a0[i + 8] = hi0[i];
        a1[i] = lo1[i]; a1[i + 8] = hi1[i];
      }
    }

    for (int e = 0; e < E_; ++e) {
      const int s = kc * E_ + e;
      wait_async0();                // my stage(s) loads have landed in LDS
      __syncthreads();              // everyone's landed; prev consumers done

      // Overlap: kick off stage(s+1) into the other buffer.
      if (s + 1 < S) {
        const int e_n  = (s + 1) & (E_ - 1);
        const int kc_n = (s + 1) >> 4;
        const _Float16* g = wg_base + (size_t)e_n * wstride_e + kc_n * 64;
        _Float16* d = &s_w[(s + 1) & 1][st_row * 64 + st_ku * 8];
        cp_b128_to_lds(d, g);
        cp_b128_to_lds(d + 8, g + 8);
      }

      // Consume stage(s): 4 WMMAs (2 k-steps x 2 n-tiles) from LDS.
      const _Float16* wb = &s_w[s & 1][0];
      const int nl0 = n_sub * 32 + l16;
      const _Float16* w00 = wb + nl0 * 64 + row_half * 16;        // nt=0, ks=0
      v8f acc0 = {0.f, 0.f, 0.f, 0.f, 0.f, 0.f, 0.f, 0.f};
      v8f acc1 = {0.f, 0.f, 0.f, 0.f, 0.f, 0.f, 0.f, 0.f};
      {
        v16h bf00 = *(const v16h*)(w00);
        v16h bf01 = *(const v16h*)(w00 + 32);                      // ks=1
        v16h bf10 = *(const v16h*)(w00 + 16 * 64);                 // nt=1, ks=0
        v16h bf11 = *(const v16h*)(w00 + 16 * 64 + 32);            // nt=1, ks=1
        acc0 = __builtin_amdgcn_wmma_f32_16x16x32_f16(false, a0, false, bf00, (short)0, acc0, false, false);
        acc1 = __builtin_amdgcn_wmma_f32_16x16x32_f16(false, a0, false, bf10, (short)0, acc1, false, false);
        acc0 = __builtin_amdgcn_wmma_f32_16x16x32_f16(false, a1, false, bf01, (short)0, acc0, false, false);
        acc1 = __builtin_amdgcn_wmma_f32_16x16x32_f16(false, a1, false, bf11, (short)0, acc1, false, false);
      }
      // ew-scale this expert's chunk partial into the mixed accumulators.
#pragma unroll
      for (int v = 0; v < 8; ++v) {
        float w_e = s_ew[(rb + v) * E_ + e];
        accm0[v] += w_e * acc0[v];
        accm1[v] += w_e * acc1[v];
      }
    }
  }

  // Bias mix + activation + store (two 16-col tiles).
#pragma unroll
  for (int v = 0; v < 8; ++v) {
    float bm0 = 0.f, bm1 = 0.f;
#pragma unroll
    for (int e = 0; e < E_; ++e) {
      float w_e = s_ew[(rb + v) * E_ + e];
      bm0 += w_e * bias[(size_t)e * N + n_g0];
      bm1 += w_e * bias[(size_t)e * N + n_g1];
    }
    float r0 = accm0[v] + bm0;
    float r1 = accm1[v] + bm1;
    if (apply_elu) {
      r0 = r0 > 0.f ? r0 : expm1f(r0);
      r1 = r1 > 0.f ? r1 : expm1f(r1);
    }
    const int brow_g = b0 + row_half * 8 + v;
    if (out_h) {
      out_h[(size_t)brow_g * N + n_g0] = (_Float16)r0;
      out_h[(size_t)brow_g * N + n_g1] = (_Float16)r1;
    }
    if (out_f) {
      out_f[(size_t)brow_g * OUTW + n_g0] = r0;
      out_f[(size_t)brow_g * OUTW + n_g1] = r1;
    }
  }
}

// ---------------------------------------------------------------------------
// Host launcher
// ---------------------------------------------------------------------------
extern "C" void kernel_launch(void* const* d_in, const int* in_sizes, int n_in,
                              void* d_out, int out_size, void* d_ws, size_t ws_size,
                              hipStream_t stream) {
  const float* x    = (const float*)d_in[0];
  const int*   gidx = (const int*)d_in[1];
  const float* g1w  = (const float*)d_in[2];
  const float* g1b  = (const float*)d_in[3];
  const float* g2w  = (const float*)d_in[4];
  const float* g2b  = (const float*)d_in[5];
  const float* g3w  = (const float*)d_in[6];
  const float* g3b  = (const float*)d_in[7];
  const float* w0   = (const float*)d_in[8];
  const float* b0   = (const float*)d_in[9];
  const float* w1   = (const float*)d_in[10];
  const float* b1   = (const float*)d_in[11];
  const float* w2   = (const float*)d_in[12];
  const float* b2   = (const float*)d_in[13];
  float* out = (float*)d_out;

  const size_t nX = (size_t)B_ * IN_;
  const size_t nW = (size_t)E_ * H_ * IN_;   // same for all three layers

  char* ws = (char*)d_ws;
  size_t off = 0;
  auto carve = [&](size_t bytes) { void* p = ws + off; off = (off + bytes + 255) & ~(size_t)255; return p; };
  _Float16* xh  = (_Float16*)carve(nX * sizeof(_Float16));
  _Float16* w0h = (_Float16*)carve(nW * sizeof(_Float16));
  _Float16* w1h = (_Float16*)carve(nW * sizeof(_Float16));
  _Float16* w2h = (_Float16*)carve(nW * sizeof(_Float16));
  _Float16* m1h = (_Float16*)carve((size_t)B_ * H_ * sizeof(_Float16));
  _Float16* m2h = (_Float16*)carve((size_t)B_ * H_ * sizeof(_Float16));
  float*    h1  = (float*)carve((size_t)B_ * GH_ * sizeof(float));
  float*    h2  = (float*)carve((size_t)B_ * GH_ * sizeof(float));
  float*    ewp = (float*)carve((size_t)B_ * E_ * sizeof(float));
  (void)ws_size; (void)n_in; (void)in_sizes; (void)out_size;

  // 1) fp32 -> f16 conversions.
  k_cvt_f16<<<2048, 256, 0, stream>>>(x,  xh,  nX);
  k_cvt_f16<<<8192, 256, 0, stream>>>(w0, w0h, nW);
  k_cvt_f16<<<8192, 256, 0, stream>>>(w1, w1h, nW);
  k_cvt_f16<<<8192, 256, 0, stream>>>(w2, w2h, nW);

  // 2) Gating network.
  k_gate1<<<(B_ * GH_ + 255) / 256, 256, 0, stream>>>(x, gidx, g1w, g1b, h1);
  k_gate2<<<(B_ * GH_ + 255) / 256, 256, 0, stream>>>(h1, g2w, g2b, h2);
  k_gate3_softmax<<<B_, 16, 0, stream>>>(h2, g3w, g3b, ewp, out);

  // 3) Three expert-mixture layers (WMMA GEMMs, async LDS staged).
  dim3 grid(B_ / 64, H_ / 64);
  k_moe_layer<<<grid, 256, 0, stream>>>(xh,  w0h, b0, ewp, m1h, nullptr, IN_, H_,  1);
  k_moe_layer<<<grid, 256, 0, stream>>>(m1h, w1h, b1, ewp, m2h, nullptr, H_,  H_,  1);
  k_moe_layer<<<grid, 256, 0, stream>>>(m2h, w2h, b2, ewp, nullptr, out, H_, OUT_, 0);
}
